// DecoderCacheModel_25451976196641
// MI455X (gfx1250) — matrix operand order repaired and checked by reading, not compiled, not run on hardware
//
#include <hip/hip_runtime.h>
#include <hip/hip_bf16.h>

// ---------------------------------------------------------------------------
// Problem constants (from setup_inputs)
// ---------------------------------------------------------------------------
#define B_   4
#define S_   2048
#define D_   1024
#define DC_  256
#define NL_  768
#define KW_  8
#define KS_  5
#define EPSF 1e-6f

// WMMA tile config: block = 256 threads = 8 waves (4 M x 2 N wave grid)
// block tile 128x128, wave tile 32x64 (2x4 of 16x16x32 bf16 WMMA)
#define BM 128
#define BN 128
#define BK 32
#define LDAS 36           // LDS row stride in bf16 elems (BK+4: 8B-aligned, conflict-skewed)
#define LDAS32 18         // same, in packed uint units
#define LDBS 36
#define LDBS32 18

typedef __attribute__((ext_vector_type(16))) __bf16 bf16x16;
typedef __attribute__((ext_vector_type(8)))  float  floatx8;

union FragBF { unsigned u[8]; bf16x16 v; };

// pack two f32 -> packed bf16 pair (lowers to v_cvt_pk_bf16_f32)
static __device__ __forceinline__ unsigned pk_bf16(float a, float b)
{
    __bf16 lo = (__bf16)a, hi = (__bf16)b;
    return (unsigned)__builtin_bit_cast(unsigned short, lo) |
           ((unsigned)__builtin_bit_cast(unsigned short, hi) << 16);
}

// A-fragment packed-uint K map used here for V_WMMA_F32_16X16X32_BF16 (contiguous, not ISA 7.12.2 order):
// vgpr j -> K=2j+16h (same order as B) => uint index = j + 8h
// B-fragment: vgpr j -> K=2j+16h                => uint index = j + 8h

// ---------------------------------------------------------------------------
// Generic batched GEMM: C[b] = alpha * opA(A[b]) @ opB(B[b])
//   transA==0: A is M x K (lda);  transA==1: A is K x M (A[k*lda+m])
//   transB==0: B is K x N (ldb);  transB==1: B is N x K (B[n*ldb+k])
// ALIGNED: M%BM==0, N%BN==0, K%BK==0, lda/ldb mult of 4 -> no bound checks.
// ---------------------------------------------------------------------------
template <bool ALIGNED>
__global__ __launch_bounds__(256) void wmma_gemm_kernel(
    const float* __restrict__ A, long long sA, int lda, int transA,
    const float* __restrict__ Bm, long long sB, int ldb, int transB,
    float* __restrict__ C, long long sC, int ldc,
    int M, int N, int K, float alpha)
{
    __shared__ __align__(16) __bf16 As[BM * LDAS];
    __shared__ __align__(16) __bf16 Bs[BN * LDBS];
    unsigned* As32 = (unsigned*)As;
    unsigned* Bs32 = (unsigned*)Bs;

    const int b = blockIdx.z;
    A  += (long long)b * sA;
    Bm += (long long)b * sB;
    C  += (long long)b * sC;

    const int tid  = threadIdx.x;
    const int lane = tid & 31;
    const int wave = tid >> 5;
    const int wm   = wave >> 1;   // 0..3 -> M offset wm*32
    const int wn   = wave & 1;    // 0..1 -> N offset wn*64
    const int row0 = blockIdx.y * BM;
    const int col0 = blockIdx.x * BN;
    const int h    = lane >> 4;
    const int mr   = lane & 15;

    // staging coordinates
    const int ar  = tid >> 1;               // A: row 0..127
    const int ac  = (tid & 1) * 16;         // A: k sub 0|16
    const int bk2 = (tid >> 4) * 2;         // B(noT): k pair base 0..30
    const int bn8 = (tid & 15) * 8;         // B(noT): n base 0..120
    const int tn  = tid >> 1;               // B(T): n 0..127
    const int tk  = (tid & 1) * 16;         // B(T): k sub 0|16

    floatx8 acc[2][4] = {};

    for (int k0 = 0; k0 < K; k0 += BK) {
        // ================= stage A tile (BM x BK) =================
        {
            const int abase = ar * LDAS32 + (ac >> 1);
            if (!transA) {
                const float* src = A + (long long)(row0 + ar) * lda + (k0 + ac);
                if (ALIGNED || ((row0 + ar) < M && (k0 + ac + 16) <= K && (lda & 3) == 0)) {
                    const float4* s4 = (const float4*)src;
                    float4 v0 = s4[0], v1 = s4[1], v2 = s4[2], v3 = s4[3];
                    As32[abase + 0] = pk_bf16(v0.x, v0.y);
                    As32[abase + 1] = pk_bf16(v0.z, v0.w);
                    As32[abase + 2] = pk_bf16(v1.x, v1.y);
                    As32[abase + 3] = pk_bf16(v1.z, v1.w);
                    As32[abase + 4] = pk_bf16(v2.x, v2.y);
                    As32[abase + 5] = pk_bf16(v2.z, v2.w);
                    As32[abase + 6] = pk_bf16(v3.x, v3.y);
                    As32[abase + 7] = pk_bf16(v3.z, v3.w);
                } else {
                    const bool rk = (row0 + ar) < M;
#pragma unroll
                    for (int u = 0; u < 16; u += 2) {
                        float v0 = (rk && (k0 + ac + u)     < K) ? src[u]     : 0.0f;
                        float v1 = (rk && (k0 + ac + u + 1) < K) ? src[u + 1] : 0.0f;
                        As32[abase + (u >> 1)] = pk_bf16(v0, v1);
                    }
                }
            } else {                                   // A[k][m]
                const int gm = row0 + ar;
                if (ALIGNED) {
#pragma unroll
                    for (int u = 0; u < 16; u += 2) {
                        float v0 = A[(long long)(k0 + ac + u)     * lda + gm];
                        float v1 = A[(long long)(k0 + ac + u + 1) * lda + gm];
                        As32[abase + (u >> 1)] = pk_bf16(v0, v1);
                    }
                } else {
                    const bool rk = gm < M;
#pragma unroll
                    for (int u = 0; u < 16; u += 2) {
                        float v0 = (rk && (k0 + ac + u)     < K) ? A[(long long)(k0 + ac + u)     * lda + gm] : 0.0f;
                        float v1 = (rk && (k0 + ac + u + 1) < K) ? A[(long long)(k0 + ac + u + 1) * lda + gm] : 0.0f;
                        As32[abase + (u >> 1)] = pk_bf16(v0, v1);
                    }
                }
            }
        }
        // ================= stage B tile (BK x BN) as [n][k] =================
        if (!transB) {                                 // B[k][n]
            const int gk = k0 + bk2;
            const float* r0p = Bm + (long long)gk * ldb + (col0 + bn8);
            if (ALIGNED || ((gk + 1) < K && (col0 + bn8 + 8) <= N && (ldb & 3) == 0)) {
                const float4* p0 = (const float4*)r0p;
                const float4* p1 = (const float4*)(r0p + ldb);
                float4 x0 = p0[0], x1 = p0[1];
                float4 y0 = p1[0], y1 = p1[1];
                const int kh = bk2 >> 1;
                Bs32[(bn8 + 0) * LDBS32 + kh] = pk_bf16(x0.x, y0.x);
                Bs32[(bn8 + 1) * LDBS32 + kh] = pk_bf16(x0.y, y0.y);
                Bs32[(bn8 + 2) * LDBS32 + kh] = pk_bf16(x0.z, y0.z);
                Bs32[(bn8 + 3) * LDBS32 + kh] = pk_bf16(x0.w, y0.w);
                Bs32[(bn8 + 4) * LDBS32 + kh] = pk_bf16(x1.x, y1.x);
                Bs32[(bn8 + 5) * LDBS32 + kh] = pk_bf16(x1.y, y1.y);
                Bs32[(bn8 + 6) * LDBS32 + kh] = pk_bf16(x1.z, y1.z);
                Bs32[(bn8 + 7) * LDBS32 + kh] = pk_bf16(x1.w, y1.w);
            } else {
                const int kh = bk2 >> 1;
#pragma unroll
                for (int u = 0; u < 8; ++u) {
                    const int gn = col0 + bn8 + u;
                    float v0 = (gk     < K && gn < N) ? Bm[(long long)gk * ldb + gn]       : 0.0f;
                    float v1 = (gk + 1 < K && gn < N) ? Bm[(long long)(gk + 1) * ldb + gn] : 0.0f;
                    Bs32[(bn8 + u) * LDBS32 + kh] = pk_bf16(v0, v1);
                }
            }
        } else {                                       // B[n][k]
            const int gn = col0 + tn;
            const float* src = Bm + (long long)gn * ldb + (k0 + tk);
            const int bbase = tn * LDBS32 + (tk >> 1);
            if (ALIGNED || (gn < N && (k0 + tk + 16) <= K && (ldb & 3) == 0)) {
                const float4* s4 = (const float4*)src;
                float4 v0 = s4[0], v1 = s4[1], v2 = s4[2], v3 = s4[3];
                Bs32[bbase + 0] = pk_bf16(v0.x, v0.y);
                Bs32[bbase + 1] = pk_bf16(v0.z, v0.w);
                Bs32[bbase + 2] = pk_bf16(v1.x, v1.y);
                Bs32[bbase + 3] = pk_bf16(v1.z, v1.w);
                Bs32[bbase + 4] = pk_bf16(v2.x, v2.y);
                Bs32[bbase + 5] = pk_bf16(v2.z, v2.w);
                Bs32[bbase + 6] = pk_bf16(v3.x, v3.y);
                Bs32[bbase + 7] = pk_bf16(v3.z, v3.w);
            } else {
                const bool nok = gn < N;
#pragma unroll
                for (int u = 0; u < 16; u += 2) {
                    float v0 = (nok && (k0 + tk + u)     < K) ? src[u]     : 0.0f;
                    float v1 = (nok && (k0 + tk + u + 1) < K) ? src[u + 1] : 0.0f;
                    Bs32[bbase + (u >> 1)] = pk_bf16(v0, v1);
                }
            }
        }
        __syncthreads();

        // ================= fragments + 2x4 WMMA =================
        FragBF fa[2], fb[4];
        const unsigned* Ac = (const unsigned*)As;
        const unsigned* Bc = (const unsigned*)Bs;
        const int akm[8] = {0, 1, 2, 3, 8, 9, 10, 11};
#pragma unroll
        for (int i = 0; i < 2; ++i) {
            const int base = (wm * 32 + i * 16 + mr) * LDAS32 + 8 * h;
#pragma unroll
            for (int j = 0; j < 8; ++j) fa[i].u[j] = Ac[base + j];
        }
#pragma unroll
        for (int j2 = 0; j2 < 4; ++j2) {
            const int base = (wn * 64 + j2 * 16 + mr) * LDBS32 + 8 * h;
#pragma unroll
            for (int j = 0; j < 8; ++j) fb[j2].u[j] = Bc[base + j];
        }
#pragma unroll
        for (int i = 0; i < 2; ++i)
#pragma unroll
            for (int j2 = 0; j2 < 4; ++j2)
                acc[i][j2] = __builtin_amdgcn_wmma_f32_16x16x32_bf16(
                    false, fa[i].v, false, fb[j2].v, (short)0, acc[i][j2], false, false);
        __syncthreads();
    }

    // ================= epilogue (C/D layout: M = vgpr + 8h, N = lane&15) ====
#pragma unroll
    for (int i = 0; i < 2; ++i)
#pragma unroll
        for (int j2 = 0; j2 < 4; ++j2)
#pragma unroll
            for (int r = 0; r < 8; ++r) {
                const int m = row0 + wm * 32 + i * 16 + r + 8 * h;
                const int n = col0 + wn * 64 + j2 * 16 + mr;
                if (ALIGNED || (m < M && n < N))
                    C[(long long)m * ldc + n] = alpha * acc[i][j2][r];
            }
}

// ---------------------------------------------------------------------------
// Causal dilated conv as 5 row-shifted WMMA GEMM taps, fused GELU + residual:
//   Y[s,o] = X[s,o] + gelu( sum_t X[s-(4-t)*dil, :] @ W[t] + bias[o] )
// S, D multiples of tile -> only the causal left-pad needs a zero-fill test.
// ---------------------------------------------------------------------------
__global__ __launch_bounds__(256) void wmma_conv_kernel(
    const float* __restrict__ X, const float* __restrict__ W,
    const float* __restrict__ bias, float* __restrict__ Y, int dil)
{
    __shared__ __align__(16) __bf16 As[BM * LDAS];
    __shared__ __align__(16) __bf16 Bs[BN * LDBS];
    unsigned* As32 = (unsigned*)As;
    unsigned* Bs32 = (unsigned*)Bs;

    const int b = blockIdx.z;
    X += (long long)b * S_ * D_;
    Y += (long long)b * S_ * D_;

    const int tid  = threadIdx.x;
    const int lane = tid & 31;
    const int wave = tid >> 5;
    const int wm   = wave >> 1;
    const int wn   = wave & 1;
    const int row0 = blockIdx.y * BM;
    const int col0 = blockIdx.x * BN;
    const int h  = lane >> 4;
    const int mr = lane & 15;

    const int ar  = tid >> 1;
    const int ac  = (tid & 1) * 16;
    const int bk2 = (tid >> 4) * 2;
    const int bn8 = (tid & 15) * 8;

    floatx8 acc[2][4] = {};

    for (int t = 0; t < KS_; ++t) {
        const int shift = (KS_ - 1 - t) * dil;
        const float* Wt = W + (long long)t * D_ * D_;
        const int srow = row0 + ar - shift;
        for (int k0 = 0; k0 < D_; k0 += BK) {
            // stage A (causal zero pad on the left edge only)
            {
                const int abase = ar * LDAS32 + (ac >> 1);
                if (srow >= 0) {
                    const float4* s4 = (const float4*)(X + (long long)srow * D_ + (k0 + ac));
                    float4 v0 = s4[0], v1 = s4[1], v2 = s4[2], v3 = s4[3];
                    As32[abase + 0] = pk_bf16(v0.x, v0.y);
                    As32[abase + 1] = pk_bf16(v0.z, v0.w);
                    As32[abase + 2] = pk_bf16(v1.x, v1.y);
                    As32[abase + 3] = pk_bf16(v1.z, v1.w);
                    As32[abase + 4] = pk_bf16(v2.x, v2.y);
                    As32[abase + 5] = pk_bf16(v2.z, v2.w);
                    As32[abase + 6] = pk_bf16(v3.x, v3.y);
                    As32[abase + 7] = pk_bf16(v3.z, v3.w);
                } else {
#pragma unroll
                    for (int j = 0; j < 8; ++j) As32[abase + j] = 0u;
                }
            }
            // stage B: Wt[k][n] (always in bounds)
            {
                const float* r0p = Wt + (long long)(k0 + bk2) * D_ + (col0 + bn8);
                const float4* p0 = (const float4*)r0p;
                const float4* p1 = (const float4*)(r0p + D_);
                float4 x0 = p0[0], x1 = p0[1];
                float4 y0 = p1[0], y1 = p1[1];
                const int kh = bk2 >> 1;
                Bs32[(bn8 + 0) * LDBS32 + kh] = pk_bf16(x0.x, y0.x);
                Bs32[(bn8 + 1) * LDBS32 + kh] = pk_bf16(x0.y, y0.y);
                Bs32[(bn8 + 2) * LDBS32 + kh] = pk_bf16(x0.z, y0.z);
                Bs32[(bn8 + 3) * LDBS32 + kh] = pk_bf16(x0.w, y0.w);
                Bs32[(bn8 + 4) * LDBS32 + kh] = pk_bf16(x1.x, y1.x);
                Bs32[(bn8 + 5) * LDBS32 + kh] = pk_bf16(x1.y, y1.y);
                Bs32[(bn8 + 6) * LDBS32 + kh] = pk_bf16(x1.z, y1.z);
                Bs32[(bn8 + 7) * LDBS32 + kh] = pk_bf16(x1.w, y1.w);
            }
            __syncthreads();

            FragBF fa[2], fb[4];
            const unsigned* Ac = (const unsigned*)As;
            const unsigned* Bc = (const unsigned*)Bs;
            const int akm[8] = {0, 1, 2, 3, 8, 9, 10, 11};
#pragma unroll
            for (int i = 0; i < 2; ++i) {
                const int base = (wm * 32 + i * 16 + mr) * LDAS32 + 8 * h;
#pragma unroll
                for (int j = 0; j < 8; ++j) fa[i].u[j] = Ac[base + j];
            }
#pragma unroll
            for (int j2 = 0; j2 < 4; ++j2) {
                const int base = (wn * 64 + j2 * 16 + mr) * LDBS32 + 8 * h;
#pragma unroll
                for (int j = 0; j < 8; ++j) fb[j2].u[j] = Bc[base + j];
            }
#pragma unroll
            for (int i = 0; i < 2; ++i)
#pragma unroll
                for (int j2 = 0; j2 < 4; ++j2)
                    acc[i][j2] = __builtin_amdgcn_wmma_f32_16x16x32_bf16(
                        false, fa[i].v, false, fb[j2].v, (short)0, acc[i][j2], false, false);
            __syncthreads();
        }
    }

    // fused epilogue: bias + tanh-GELU + residual (always in bounds)
#pragma unroll
    for (int i = 0; i < 2; ++i)
#pragma unroll
        for (int j2 = 0; j2 < 4; ++j2)
#pragma unroll
            for (int r = 0; r < 8; ++r) {
                const int m = row0 + wm * 32 + i * 16 + r + 8 * h;
                const int n = col0 + wn * 64 + j2 * 16 + mr;
                float y = acc[i][j2][r] + bias[n];
                float g = 0.5f * y * (1.0f + tanhf(0.7978845608f * (y + 0.044715f * y * y * y)));
                Y[(long long)m * D_ + n] = X[(long long)m * D_ + n] + g;
            }
}

// ---------------------------------------------------------------------------
// Row-wise helpers (one 256-thread block per row)
// ---------------------------------------------------------------------------
__global__ __launch_bounds__(256) void softmax_rows(float* __restrict__ data, int N)
{
    __shared__ float red[256];
    float* row = data + (long long)blockIdx.x * N;
    const int tid = threadIdx.x;

    float mx = -3.4e38f;
    for (int i = tid; i < N; i += 256) mx = fmaxf(mx, row[i]);
    red[tid] = mx; __syncthreads();
    for (int s = 128; s > 0; s >>= 1) { if (tid < s) red[tid] = fmaxf(red[tid], red[tid + s]); __syncthreads(); }
    mx = red[0]; __syncthreads();

    float sum = 0.0f;
    for (int i = tid; i < N; i += 256) { float e = expf(row[i] - mx); row[i] = e; sum += e; }
    red[tid] = sum; __syncthreads();
    for (int s = 128; s > 0; s >>= 1) { if (tid < s) red[tid] += red[tid + s]; __syncthreads(); }
    const float inv = 1.0f / red[0];
    for (int i = tid; i < N; i += 256) row[i] *= inv;
}

// O = X + sigmoid(X . wg) * P  (per row, O may alias X)
__global__ __launch_bounds__(256) void gated_residual(
    const float* __restrict__ X, const float* __restrict__ P,
    const float* __restrict__ wg, float* __restrict__ O)
{
    __shared__ float red[256];
    const long long row = blockIdx.x;
    const float* x = X + row * D_;
    const float* p = P + row * D_;
    float* o = O + row * D_;
    const int tid = threadIdx.x;

    float d = 0.0f;
    for (int i = tid; i < D_; i += 256) d += x[i] * wg[i];
    red[tid] = d; __syncthreads();
    for (int s = 128; s > 0; s >>= 1) { if (tid < s) red[tid] += red[tid + s]; __syncthreads(); }
    const float g = 1.0f / (1.0f + expf(-red[0]));
    for (int i = tid; i < D_; i += 256) o[i] = x[i] + g * p[i];
}

// Wrow[row,k] = sigmoid(X[row] . wg) * Wrow[row,k]  (in place)
__global__ __launch_bounds__(256) void gated_weights(
    const float* __restrict__ X, const float* __restrict__ wg,
    float* __restrict__ Wr, int Kd)
{
    __shared__ float red[256];
    const long long row = blockIdx.x;
    const float* x = X + row * D_;
    float* w = Wr + row * Kd;
    const int tid = threadIdx.x;

    float d = 0.0f;
    for (int i = tid; i < D_; i += 256) d += x[i] * wg[i];
    red[tid] = d; __syncthreads();
    for (int s = 128; s > 0; s >>= 1) { if (tid < s) red[tid] += red[tid + s]; __syncthreads(); }
    const float g = 1.0f / (1.0f + expf(-red[0]));
    for (int k = tid; k < Kd; k += 256) w[k] = g * w[k];
}

// tot[b,k] = sum_s W[b,s,k]
__global__ __launch_bounds__(256) void colsum_kernel(
    const float* __restrict__ W, float* __restrict__ tot, int Kd)
{
    __shared__ float red[256];
    const int b = blockIdx.x / Kd;
    const int k = blockIdx.x % Kd;
    const int tid = threadIdx.x;
    const float* base = W + ((long long)b * S_) * Kd + k;
    float s = 0.0f;
    for (int i = tid; i < S_; i += 256) s += base[(long long)i * Kd];
    red[tid] = s; __syncthreads();
    for (int t = 128; t > 0; t >>= 1) { if (tid < t) red[tid] += red[tid + t]; __syncthreads(); }
    if (tid == 0) tot[blockIdx.x] = red[0];
}

__global__ __launch_bounds__(256) void layernorm_rows(
    const float* __restrict__ X, const float* __restrict__ g,
    const float* __restrict__ bta, float* __restrict__ O)
{
    __shared__ float r0[256];
    __shared__ float r1[256];
    const long long row = blockIdx.x;
    const float* x = X + row * D_;
    float* o = O + row * D_;
    const int tid = threadIdx.x;

    float s = 0.0f, s2 = 0.0f;
    for (int i = tid; i < D_; i += 256) { float v = x[i]; s += v; s2 += v * v; }
    r0[tid] = s; r1[tid] = s2; __syncthreads();
    for (int t = 128; t > 0; t >>= 1) {
        if (tid < t) { r0[tid] += r0[tid + t]; r1[tid] += r1[tid + t]; }
        __syncthreads();
    }
    const float mean = r0[0] / (float)D_;
    const float var  = r1[0] / (float)D_ - mean * mean;
    const float inv  = rsqrtf(var + 1e-5f);
    for (int i = tid; i < D_; i += 256)
        o[i] = (x[i] - mean) * inv * g[i] + bta[i];
}

// new_wm[b,k,:DC] = (1-a)*content + a*raw/(tot+eps); new_wm[b,k,DC]=clip(valid+tot)
__global__ __launch_bounds__(256) void finalize_wm(
    const float* __restrict__ wm, const float* __restrict__ raw,
    const float* __restrict__ tot, float* __restrict__ outwm)
{
    const int b = blockIdx.x / KW_;
    const int k = blockIdx.x % KW_;
    const int d = threadIdx.x;                 // 0..255 == DC_
    const float t = tot[b * KW_ + k];
    const float a = fminf(fmaxf(t, 0.0f), 1.0f);
    const long long base = ((long long)b * KW_ + k) * (DC_ + 1);
    const float content = wm[base + d];
    const float avg = raw[((long long)b * KW_ + k) * DC_ + d] / (t + EPSF);
    outwm[base + d] = (1.0f - a) * content + a * avg;
    if (d == 0)
        outwm[base + DC_] = fminf(fmaxf(wm[base + DC_] + t, 0.0f), 1.0f);
}

__global__ __launch_bounds__(256) void finalize_cache(
    const float* __restrict__ cache, const float* __restrict__ raw,
    const float* __restrict__ tot, float* __restrict__ outc)
{
    const int b = blockIdx.x / NL_;
    const int n = blockIdx.x % NL_;
    const int d = threadIdx.x;                 // 0..255 == DC_
    const float t = tot[b * NL_ + n];
    const float a = fminf(fmaxf(t, 0.0f), 1.0f);
    const long long base = ((long long)b * NL_ + n) * DC_;
    const float avg = raw[base + d] / (t + EPSF);
    outc[base + d] = (1.0f - a) * cache[base + d] + a * avg;
}

// ---------------------------------------------------------------------------
// Host-side orchestration
// ---------------------------------------------------------------------------
static inline void gemm(hipStream_t st,
    const float* A, long long sA, int lda, int tA,
    const float* Bm, long long sB, int ldb, int tB,
    float* C, long long sC, int ldc,
    int M, int N, int K, int batch, float alpha)
{
    dim3 grid((N + BN - 1) / BN, (M + BM - 1) / BM, batch);
    const bool aligned = (M % BM == 0) && (N % BN == 0) && (K % BK == 0) &&
                         ((lda & 3) == 0) && ((ldb & 3) == 0);
    if (aligned)
        wmma_gemm_kernel<true><<<grid, dim3(256), 0, st>>>(A, sA, lda, tA, Bm, sB, ldb, tB,
                                                           C, sC, ldc, M, N, K, alpha);
    else
        wmma_gemm_kernel<false><<<grid, dim3(256), 0, st>>>(A, sA, lda, tA, Bm, sB, ldb, tB,
                                                            C, sC, ldc, M, N, K, alpha);
}

extern "C" void kernel_launch(void* const* d_in, const int* in_sizes, int n_in,
                              void* d_out, int out_size, void* d_ws, size_t ws_size,
                              hipStream_t stream)
{
    (void)in_sizes; (void)n_in; (void)out_size; (void)ws_size;

    const float* x        = (const float*)d_in[0];
    const float* cache    = (const float*)d_in[1];
    const float* wm       = (const float*)d_in[2];
    const float* Wq_ltm   = (const float*)d_in[3];
    const float* Wo_ltm   = (const float*)d_in[4];
    const float* wg_ltm   = (const float*)d_in[5];
    const float* Wq_wm    = (const float*)d_in[6];
    const float* Wo_wm    = (const float*)d_in[7];
    const float* wg_wm    = (const float*)d_in[8];
    const float* conv1_w  = (const float*)d_in[9];
    const float* conv1_b  = (const float*)d_in[10];
    const float* conv2_w  = (const float*)d_in[11];
    const float* conv2_b  = (const float*)d_in[12];
    const float* ln_g     = (const float*)d_in[13];
    const float* ln_b     = (const float*)d_in[14];
    const float* Wqw_wm   = (const float*)d_in[15];
    const float* Ww_wm    = (const float*)d_in[16];
    const float* wm_keys  = (const float*)d_in[17];
    const float* wgw_wm   = (const float*)d_in[18];
    const float* Wqw_ltm  = (const float*)d_in[19];
    const float* Ww_ltm   = (const float*)d_in[20];
    const float* wgw_ltm  = (const float*)d_in[21];

    const long long BS = (long long)B_ * S_;
    float* out_main  = (float*)d_out;                         // (B,S,D)
    float* out_cache = out_main + BS * D_;                    // (B,NL,DC)
    float* out_wm    = out_cache + (long long)B_ * NL_ * DC_; // (B,KW,DC+1)

    // workspace layout (floats)
    float* ws = (float*)d_ws;
    float* t_bsd0 = ws;                           // B*S*D
    float* t_bsd1 = t_bsd0 + BS * D_;             // B*S*D
    float* t_q    = t_bsd1 + BS * D_;             // B*S*DC
    float* t_read = t_q    + BS * DC_;            // B*S*DC
    float* t_c    = t_read + BS * DC_;            // B*S*DC
    float* t_big  = t_c    + BS * DC_;            // B*S*NL  (scores / wl)
    float* t_sm0  = t_big  + BS * NL_;            // B*S*KW
    float* t_tot  = t_sm0  + BS * KW_;            // B*NL (also used for B*KW)
    float* t_raw  = t_tot  + (long long)B_ * NL_; // B*NL*DC (also B*KW*DC)

    const float inv_sqrt = 0.0625f;               // 1/sqrt(DC)
    const long long sXD = (long long)S_ * D_;
    const long long sXC = (long long)S_ * DC_;

    // ---- LTM read attention -------------------------------------------------
    gemm(stream, x, sXD, D_, 0, Wq_ltm, 0, DC_, 0, t_q, sXC, DC_, S_, DC_, D_, B_, 1.0f);
    gemm(stream, t_q, sXC, DC_, 0, cache, (long long)NL_ * DC_, DC_, 1,
         t_big, (long long)S_ * NL_, NL_, S_, NL_, DC_, B_, inv_sqrt);
    softmax_rows<<<dim3(B_ * S_), dim3(256), 0, stream>>>(t_big, NL_);
    gemm(stream, t_big, (long long)S_ * NL_, NL_, 0, cache, (long long)NL_ * DC_, DC_, 0,
         t_read, sXC, DC_, S_, DC_, NL_, B_, 1.0f);
    gemm(stream, t_read, sXC, DC_, 0, Wo_ltm, 0, D_, 0, t_bsd0, sXD, D_, S_, D_, DC_, B_, 1.0f);
    gated_residual<<<dim3(B_ * S_), dim3(256), 0, stream>>>(x, t_bsd0, wg_ltm, t_bsd1);

    // ---- WM read attention (content rows have stride DC+1) -----------------
    gemm(stream, t_bsd1, sXD, D_, 0, Wq_wm, 0, DC_, 0, t_q, sXC, DC_, S_, DC_, D_, B_, 1.0f);
    gemm(stream, t_q, sXC, DC_, 0, wm, (long long)KW_ * (DC_ + 1), DC_ + 1, 1,
         t_sm0, (long long)S_ * KW_, KW_, S_, KW_, DC_, B_, inv_sqrt);
    softmax_rows<<<dim3(B_ * S_), dim3(256), 0, stream>>>(t_sm0, KW_);
    gemm(stream, t_sm0, (long long)S_ * KW_, KW_, 0, wm, (long long)KW_ * (DC_ + 1), DC_ + 1, 0,
         t_read, sXC, DC_, S_, DC_, KW_, B_, 1.0f);
    gemm(stream, t_read, sXC, DC_, 0, Wo_wm, 0, D_, 0, t_bsd0, sXD, D_, S_, D_, DC_, B_, 1.0f);
    gated_residual<<<dim3(B_ * S_), dim3(256), 0, stream>>>(t_bsd1, t_bsd0, wg_wm, t_bsd1);

    // ---- causal dilated convs + layernorm ----------------------------------
    dim3 cgrid(D_ / BN, (S_ + BM - 1) / BM, B_);
    wmma_conv_kernel<<<cgrid, dim3(256), 0, stream>>>(t_bsd1, conv1_w, conv1_b, t_bsd0, 1);
    wmma_conv_kernel<<<cgrid, dim3(256), 0, stream>>>(t_bsd0, conv2_w, conv2_b, t_bsd1, 2);
    layernorm_rows<<<dim3(B_ * S_), dim3(256), 0, stream>>>(t_bsd1, ln_g, ln_b, out_main);

    // ---- WM write-back ------------------------------------------------------
    gemm(stream, out_main, sXD, D_, 0, Wqw_wm, 0, DC_, 0, t_q, sXC, DC_, S_, DC_, D_, B_, 1.0f);
    gemm(stream, t_q, sXC, DC_, 0, wm_keys, 0, DC_, 1,
         t_sm0, (long long)S_ * KW_, KW_, S_, KW_, DC_, B_, inv_sqrt);
    softmax_rows<<<dim3(B_ * S_), dim3(256), 0, stream>>>(t_sm0, KW_);
    gated_weights<<<dim3(B_ * S_), dim3(256), 0, stream>>>(out_main, wgw_wm, t_sm0, KW_);
    colsum_kernel<<<dim3(B_ * KW_), dim3(256), 0, stream>>>(t_sm0, t_tot, KW_);
    gemm(stream, out_main, sXD, D_, 0, Ww_wm, 0, DC_, 0, t_c, sXC, DC_, S_, DC_, D_, B_, 1.0f);
    // raw[b] = w^T @ c : transA, M=KW, N=DC, K=S
    gemm(stream, t_sm0, (long long)S_ * KW_, KW_, 1, t_c, sXC, DC_, 0,
         t_raw, (long long)KW_ * DC_, DC_, KW_, DC_, S_, B_, 1.0f);
    finalize_wm<<<dim3(B_ * KW_), dim3(256), 0, stream>>>(wm, t_raw, t_tot, out_wm);

    // ---- LTM write-back -----------------------------------------------------
    gemm(stream, out_main, sXD, D_, 0, Wqw_ltm, 0, DC_, 0, t_q, sXC, DC_, S_, DC_, D_, B_, 1.0f);
    gemm(stream, t_q, sXC, DC_, 0, cache, (long long)NL_ * DC_, DC_, 1,
         t_big, (long long)S_ * NL_, NL_, S_, NL_, DC_, B_, inv_sqrt);
    softmax_rows<<<dim3(B_ * S_), dim3(256), 0, stream>>>(t_big, NL_);
    gated_weights<<<dim3(B_ * S_), dim3(256), 0, stream>>>(out_main, wgw_ltm, t_big, NL_);
    colsum_kernel<<<dim3(B_ * NL_), dim3(256), 0, stream>>>(t_big, t_tot, NL_);
    gemm(stream, out_main, sXD, D_, 0, Ww_ltm, 0, DC_, 0, t_c, sXC, DC_, S_, DC_, D_, B_, 1.0f);
    // rawl[b] = wl^T @ cl : transA, M=NL, N=DC, K=S
    gemm(stream, t_big, (long long)S_ * NL_, NL_, 1, t_c, sXC, DC_, 0,
         t_raw, (long long)NL_ * DC_, DC_, NL_, DC_, S_, B_, 1.0f);
    finalize_cache<<<dim3(B_ * NL_), dim3(256), 0, stream>>>(cache, t_raw, t_tot, out_cache);
}
